// LSTMunit_53094385714012
// MI455X (gfx1250) — matrix operand order repaired and checked by reading, not compiled, not run on hardware
//
#include <hip/hip_runtime.h>
#include <stdint.h>

// ---------------------------------------------------------------------------
// ConvLSTM cell for MI455X (gfx1250): implicit-GEMM convs on v_wmma_f32_16x16x32_f16
// M=1024 (4 gates x 256 ch), N=65536 (B*16*16), K=4608 (2 branches * 256ci * 9 taps)
// ---------------------------------------------------------------------------

typedef __attribute__((ext_vector_type(16))) _Float16 v16h;
typedef __attribute__((ext_vector_type(8)))  float    v8f;
typedef __attribute__((ext_vector_type(4)))  unsigned u32x4;

#define NB    256      // batch
#define NCH   256      // channels per gate
#define NPOS  256      // 16*16 spatial
#define KTILES 144     // 9 taps * 2 branches * (256/32) ci-blocks
#define MROWS 1024     // 4*NCH

struct F32B { u32x4 a, b; };          // 32 bytes == one v16h fragment
struct F64B { u32x4 a, b, c, d; };    // 64 bytes == 32 f16 staged per thread

__device__ __forceinline__ v16h frag_cast(const F32B f) {
    return __builtin_bit_cast(v16h, f);
}

// ---------------------------------------------------------------------------
// Pack weights into WMMA-A-operand order.
// t = kk*16 + branch*8 + ciblk  (kk=3x3 tap, branch: 0=Wx 1=Wh, ciblk: 32-ci block)
// slot s = h*16 + e maps to K = h*16+e   (contiguous order, matches B staging)
// Layout Apack[t][co][s] so a wave's 32 lane-loads are 1KB contiguous.
// ---------------------------------------------------------------------------
__global__ void pack_w_kernel(const float* __restrict__ Wx,
                              const float* __restrict__ Wh,
                              _Float16* __restrict__ Apack) {
    unsigned idx = blockIdx.x * 256u + threadIdx.x;
    if (idx >= (unsigned)(KTILES * MROWS * 32)) return;
    int s  = idx & 31;
    int co = (idx >> 5) & 1023;
    int t  = idx >> 15;               // 1024*32 = 32768 per tile
    int kk     = t >> 4;
    int branch = (t >> 3) & 1;
    int cib    = (t & 7) << 5;
    int h = s >> 4, e = s & 15;
    int K = h * 16 + e;
    int ci = cib + K;
    const float* W = branch ? Wh : Wx;
    float v = W[(co * 256 + ci) * 9 + kk];
    Apack[idx] = (_Float16)v;
}

// ---------------------------------------------------------------------------
// Convert activations fp32 -> f16 with transpose to position-major per image:
// dst[b][pos][ci].  LDS tile transpose so both global reads (fp32 rows) and
// global writes (128B/thread f16) are fully coalesced.
// One block: 64 ci x 256 pos for one (branch, image, ci-block).
// ---------------------------------------------------------------------------
__global__ void pack_act_kernel(const float* __restrict__ x,
                                const float* __restrict__ h,
                                _Float16* __restrict__ Xp,
                                _Float16* __restrict__ Hp) {
    __shared__ _Float16 tile[256 * 72];   // row stride 72 f16 = 144B (16B aligned)
    const unsigned blk    = blockIdx.x;           // 0..2047
    const unsigned cib    = (blk & 3) * 64;       // ci block base
    const unsigned b      = (blk >> 2) & 255;
    const unsigned branch = blk >> 10;
    const float* src = branch ? h : x;
    _Float16*    dst = branch ? Hp : Xp;
    const unsigned tid = threadIdx.x;             // = position

    for (unsigned j = 0; j < 64; ++j) {           // coalesced fp32 row reads
        float v = src[((size_t)b * 256 + cib + j) * 256 + tid];
        tile[tid * 72 + j] = (_Float16)v;
    }
    __syncthreads();
    const F64B* s = (const F64B*)(tile + tid * 72);
    F64B* d = (F64B*)(dst + (size_t)b * 65536 + (size_t)tid * 256 + cib);
    d[0] = s[0];                                   // 128B contiguous per thread
    d[1] = s[1];
}

// ---------------------------------------------------------------------------
// Main fused kernel: GEMM (both convs) + bias + peephole LSTM epilogue.
// Block = 256 threads = 8 waves = (gate 0..3) x (channel half 0..1).
// Each wave: 16 rows x 256 positions accumulated in 16 v8f WMMA tiles.
// Double-buffered LDS B tile (1 barrier/K-tile); A fragment + staging regs
// double-buffered across K-tiles; B fragments dual-buffered — the asm pin
// sits AFTER the WMMA pair so the dscnt wait for the prefetched pair lands
// after the WMMAs have issued (LDS latency overlaps matrix-pipe execution).
// ---------------------------------------------------------------------------
union __align__(16) SmemU {
    _Float16 bt[2][NPOS * 40];       // double-buffered im2col tile (stride 40 f16)
    float    ep[4 * 16 * NPOS];      // epilogue gate values (one channel half)
};

__device__ __forceinline__ F64B load_stage(const _Float16* __restrict__ Xp,
                                           const _Float16* __restrict__ Hp,
                                           int b, int py, int px, int t) {
    const int kk     = t >> 4;
    const int branch = (t >> 3) & 1;
    const int cib    = (t & 7) << 5;
    const int dy = kk / 3 - 1;
    const int dx = kk - (kk / 3) * 3 - 1;
    const int yy = py + dy, xx = px + dx;
    F64B r;
    u32x4 z = {0u, 0u, 0u, 0u};
    r.a = z; r.b = z; r.c = z; r.d = z;
    if ((yy >= 0) & (yy < 16) & (xx >= 0) & (xx < 16)) {
        const _Float16* src = branch ? Hp : Xp;
        r = *(const F64B*)(src + ((size_t)b * 65536 + (yy * 16 + xx) * 256 + cib));
    }
    return r;
}

__device__ __forceinline__ v16h load_a(const _Float16* __restrict__ Apack,
                                       int t, int co, int lh) {
    return frag_cast(*(const F32B*)(Apack + ((size_t)t * 1024 + co) * 32 + lh * 16));
}

__device__ __forceinline__ v16h load_b(const _Float16* base, int nt, int ln, int lh) {
    return frag_cast(*(const F32B*)(base + (nt * 16 + ln) * 40 + lh * 16));
}

__launch_bounds__(256)
__global__ void convlstm_main_kernel(const _Float16* __restrict__ Apack,
                                     const _Float16* __restrict__ Xp,
                                     const _Float16* __restrict__ Hp,
                                     const float* __restrict__ c_in,
                                     const float* __restrict__ bh,
                                     const float* __restrict__ Wci,
                                     const float* __restrict__ Wcf,
                                     const float* __restrict__ Wco,
                                     float* __restrict__ out) {
    __shared__ SmemU smem;

    const int b     = blockIdx.x;          // image
    const int chblk = blockIdx.y;          // 32-channel block (0..7)
    const int tid   = threadIdx.x;
    const int lane  = tid & 31;
    const int wave  = tid >> 5;            // 0..7
    const int gate  = wave >> 1;           // 0..3  (i,f,c,o)
    const int half  = wave & 1;            // 0..1  (16-channel half)
    const int ln    = lane & 15;
    const int lh    = lane >> 4;

    v8f acc[16] = {};                      // 16 N-tiles of 16x16 f32

    const int p  = tid;                    // this thread's staged position
    const int py = p >> 4, px = p & 15;
    const int co = gate * 256 + chblk * 32 + half * 16 + ln;   // A row

    // ---- prologue: preload tile 0 staging + A fragment into registers ------
    F64B stage  = load_stage(Xp, Hp, b, py, px, 0);
    v16h a_cur  = load_a(Apack, 0, co, lh);

    for (int t = 0; t < KTILES; ++t) {
        // store tile t (buffer t&1: last read at t-2, safe after t-1's barrier)
        *(F64B*)(smem.bt[t & 1] + p * 40) = stage;
        __syncthreads();

        // issue next tile's global loads early so they overlap the WMMAs
        if (t + 1 < KTILES)
            stage = load_stage(Xp, Hp, b, py, px, t + 1);
        const v16h a_use = a_cur;
        if (t + 1 < KTILES) {
            a_cur = load_a(Apack, t + 1, co, lh);
            if (t + 2 < KTILES)
                __builtin_prefetch(Apack + ((size_t)(t + 2) * 1024 + co) * 32 + lh * 16, 0, 3);
        }

        // 16 WMMAs on this K-tile; two live B fragments. Pin the prefetched
        // pair AFTER the WMMAs: the dscnt wait for those loads is then placed
        // behind the issued WMMA pair instead of in front of it.
        const _Float16* base = smem.bt[t & 1];
        v16h bf0 = load_b(base, 0, ln, lh);
        v16h bf1 = load_b(base, 1, ln, lh);
#pragma unroll
        for (int nt = 0; nt < 16; nt += 2) {
            v16h n0 = bf0, n1 = bf1;
            if (nt + 2 < 16) {
                n0 = load_b(base, nt + 2, ln, lh);
                n1 = load_b(base, nt + 3, ln, lh);
            }
            acc[nt]     = __builtin_amdgcn_wmma_f32_16x16x32_f16(
                false, a_use, false, bf0, (short)0, acc[nt],     false, false);
            acc[nt + 1] = __builtin_amdgcn_wmma_f32_16x16x32_f16(
                false, a_use, false, bf1, (short)0, acc[nt + 1], false, false);
            asm("" : "+v"(n0), "+v"(n1));   // pin pair; wait lands post-WMMA
            bf0 = n0; bf1 = n1;
        }
    }

    // ---------------- epilogue: two phases (channel halves) through LDS ------
    for (int hs = 0; hs < 2; ++hs) {
        __syncthreads();
        if (half == hs) {
#pragma unroll
            for (int nt = 0; nt < 16; ++nt) {
#pragma unroll
                for (int r = 0; r < 8; ++r) {
                    const int chl = r + lh * 8;        // C/D layout: M = r + 8*laneHalf
                    const int pos = nt * 16 + ln;
                    smem.ep[(gate * 16 + chl) * NPOS + pos] = acc[nt][r];
                }
            }
        }
        __syncthreads();

        for (int e = tid; e < 16 * NPOS; e += 256) {
            const int chl = e >> 8;
            const int pos = e & 255;
            const int ch  = chblk * 32 + hs * 16 + chl;
            const float gi = smem.ep[(0 * 16 + chl) * NPOS + pos] + bh[0 * 256 + ch];
            const float gf = smem.ep[(1 * 16 + chl) * NPOS + pos] + bh[1 * 256 + ch];
            const float gc = smem.ep[(2 * 16 + chl) * NPOS + pos] + bh[2 * 256 + ch];
            const float go = smem.ep[(3 * 16 + chl) * NPOS + pos] + bh[3 * 256 + ch];
            const size_t gidx = ((size_t)b * 256 + ch) * 256 + pos;
            const int    pidx = ch * 256 + pos;        // peephole params broadcast over batch
            const float cc = c_in[gidx];
            const float it = 1.0f / (1.0f + __expf(-(gi + Wci[pidx] * cc)));
            const float ft = 1.0f / (1.0f + __expf(-(gf + Wcf[pidx] * cc)));
            const float ct = ft * cc + it * tanhf(gc);
            const float ot = 1.0f / (1.0f + __expf(-(go + Wco[pidx] * ct)));
            const float ht = ot * tanhf(ct);
            out[gidx] = ht;                             // h_t
            out[(size_t)16777216 + gidx] = ct;          // c_t
        }
    }
}

// ---------------------------------------------------------------------------
// Launch: pack weights + activations into workspace, then fused kernel.
// Workspace: Apack 9.44MB | Xp 33.55MB | Hp 33.55MB  (~76.5MB total)
// ---------------------------------------------------------------------------
extern "C" void kernel_launch(void* const* d_in, const int* in_sizes, int n_in,
                              void* d_out, int out_size, void* d_ws, size_t ws_size,
                              hipStream_t stream) {
    const float* x   = (const float*)d_in[0];
    const float* h   = (const float*)d_in[1];
    const float* c   = (const float*)d_in[2];
    const float* Wx  = (const float*)d_in[3];
    const float* Wh  = (const float*)d_in[4];
    const float* bh  = (const float*)d_in[5];
    const float* Wci = (const float*)d_in[6];
    const float* Wcf = (const float*)d_in[7];
    const float* Wco = (const float*)d_in[8];
    float* out = (float*)d_out;

    char* ws = (char*)d_ws;
    _Float16* Apack = (_Float16*)ws;                                  // 9,437,184 B
    _Float16* Xp    = (_Float16*)(ws + 9437184);                      // 33,554,432 B
    _Float16* Hp    = (_Float16*)(ws + 9437184 + 33554432);           // 33,554,432 B

    pack_w_kernel<<<(KTILES * MROWS * 32) / 256, 256, 0, stream>>>(Wx, Wh, Apack);
    pack_act_kernel<<<2048, 256, 0, stream>>>(x, h, Xp, Hp);

    dim3 grid(NB, 8, 1);   // 256 images x 8 channel-blocks
    convlstm_main_kernel<<<grid, 256, 0, stream>>>(Apack, Xp, Hp, c, bh,
                                                   Wci, Wcf, Wco, out);
}
